// splineconv_41480794145015
// MI455X (gfx1250) — compile-verified, matrix operand order checked by reading, not run on hardware
//
#include <hip/hip_runtime.h>
#include <hip/hip_bf16.h>

// ---------------------------------------------------------------------------
// SplineConv (dim=1, degree=1, open spline, mean aggr) x2 + log_softmax
// for MI455X / gfx1250: bf16 WMMA transforms, fp32 atomic aggregation.
// ---------------------------------------------------------------------------

typedef __attribute__((ext_vector_type(16))) __bf16 v16bf;
typedef __attribute__((ext_vector_type(8)))  __bf16 v8bf;
typedef __attribute__((ext_vector_type(4)))  __bf16 v4bf;
typedef __attribute__((ext_vector_type(2)))  __bf16 v2bf;
typedef __attribute__((ext_vector_type(8)))  float  v8f;

#define KS 8            // spline kernel size
#define NSLOT 9         // 8 spline kernels + 1 root slot

template <int CPL> struct bfvec;
template <> struct bfvec<4> { using type = v4bf; };
template <> struct bfvec<2> { using type = v2bf; };

// ---------------- fp32 -> bf16 convert ----------------
__global__ void k_cvt_bf16(const float* __restrict__ src,
                           __bf16* __restrict__ dst, int n) {
  int i = blockIdx.x * blockDim.x + threadIdx.x;
  if (i < n) dst[i] = (__bf16)src[i];
}

// ---------------- pack W (+root) into WMMA B-fragment layout ---------------
// Wp layout: [slot][colTile][chunk][lane(32)][16 bf16]
// Lane (half = lane>>4, col = colTile*16 + (lane&15)) element j corresponds to
// K = chunk*32 + (j<8 ? half*8+j : 16 + half*8 + (j-8))   (ISA 16-bit B tile)
template <int CIN, int COUTP, int COUTR>
__global__ void k_pack_w(const float* __restrict__ W,     // [NSLOT-1][CIN][COUTR]
                         const float* __restrict__ root,  // [CIN][COUTR]
                         __bf16* __restrict__ Wp) {
  constexpr int colTiles = COUTP / 16, chunks = CIN / 32;
  int idx = blockIdx.x * blockDim.x + threadIdx.x;
  int total = NSLOT * colTiles * chunks * 32;
  if (idx >= total) return;
  int lane = idx & 31;
  int t = idx >> 5;
  int chunk = t % chunks; t /= chunks;
  int colTile = t % colTiles;
  int slot = t / colTiles;
  int half = lane >> 4;
  int col = colTile * 16 + (lane & 15);
  const float* src = (slot == NSLOT - 1) ? root : (W + (size_t)slot * CIN * COUTR);
  __bf16* out = Wp + (size_t)idx * 16;
#pragma unroll
  for (int j = 0; j < 16; ++j) {
    int kk = chunk * 32 + ((j < 8) ? (half * 8 + j) : (16 + half * 8 + (j - 8)));
    float w = (col < COUTR) ? src[(size_t)kk * COUTR + col] : 0.0f;
    out[j] = (__bf16)w;
  }
}

// ---------------- WMMA transform: T[n,slot,c] = Xb[n,:] @ Wslot ------------
// One wave per (rowTile, slot): A fragments held in VGPRs across all column
// tiles; B tiles streamed from the packed (L2-resident) weight table.
template <int CIN, int COUTP>
__global__ __launch_bounds__(256)
void k_transform(const __bf16* __restrict__ Xb,   // [nrows][CIN]
                 const __bf16* __restrict__ Wp,   // packed B fragments
                 __bf16* __restrict__ T,          // [nrows][NSLOT][COUTP]
                 int nrows, int rowTiles) {
  constexpr int colTiles = COUTP / 16, chunks = CIN / 32;
  int wave = (int)((blockIdx.x * blockDim.x + threadIdx.x) >> 5);
  int lane = threadIdx.x & 31;
  int total = rowTiles * NSLOT;
  if (wave >= total) return;
  int slot = wave % NSLOT;
  int rowTile = wave / NSLOT;
  int half = lane >> 4;
  int l15 = lane & 15;
  int rowBase = rowTile * 16;
  bool full = (rowBase + 16) <= nrows;
  int row = rowBase + l15;
  if (row >= nrows) row = nrows - 1;          // clamp (stores are guarded)
  const __bf16* arow = Xb + (size_t)row * CIN;

  // A fragments (16-bit 16x32 layout): lanes 0-15 K=[0..8)+[16..24),
  // lanes 16-31 K=[8..16)+[24..32) within each K=32 chunk.
  v16bf A[chunks];
#pragma unroll
  for (int c = 0; c < chunks; ++c) {
    v8bf alo = *(const v8bf*)(arow + c * 32 + half * 8);
    v8bf ahi = *(const v8bf*)(arow + c * 32 + 16 + half * 8);
#pragma unroll
    for (int j = 0; j < 8; ++j) { A[c][j] = alo[j]; A[c][j + 8] = ahi[j]; }
  }

  const __bf16* bslot = Wp + (size_t)slot * colTiles * chunks * 32 * 16;
  constexpr size_t rstride = (size_t)NSLOT * COUTP;  // elements between rows
  // base pointer for this lane's D rows (VGPR r -> row half?8+r:r)
  __bf16* outBase =
      T + ((size_t)(rowBase + (half ? 8 : 0)) * NSLOT + slot) * COUTP + l15;

#pragma unroll
  for (int ct = 0; ct < colTiles; ++ct) {
    v8f acc = {0.f, 0.f, 0.f, 0.f, 0.f, 0.f, 0.f, 0.f};
#pragma unroll
    for (int c = 0; c < chunks; ++c) {
      v16bf B = *(const v16bf*)(bslot + ((size_t)(ct * chunks + c) * 32 + lane) * 16);
      acc = __builtin_amdgcn_wmma_f32_16x16x32_bf16(
          /*neg_a=*/false, A[c], /*neg_b=*/false, B,
          /*c_mod=*/(short)0, acc, /*reuse_a=*/false, /*reuse_b=*/false);
    }
    __bf16* outp = outBase + ct * 16;
    if (full) {
#pragma unroll
      for (int r = 0; r < 8; ++r) outp[(size_t)r * rstride] = (__bf16)acc[r];
    } else {
#pragma unroll
      for (int r = 0; r < 8; ++r) {
        int rr = rowBase + (half ? 8 + r : r);
        if (rr < nrows) outp[(size_t)r * rstride] = (__bf16)acc[r];
      }
    }
  }
}

// ---------------- degree (incoming edge count) ----------------
__global__ void k_degree(const long long* __restrict__ ei,
                         float* __restrict__ deg, int E) {
  int e = blockIdx.x * blockDim.x + threadIdx.x;
  if (e < E) atomicAdd(&deg[(int)ei[(size_t)E + e]], 1.0f);
}

// ---------------- per-edge gather / spline blend / scatter-add -------------
// One wave per edge; each lane handles CPL contiguous channels.
template <int COUTP, int COUTA, int CPL>
__global__ __launch_bounds__(256)
void k_edge(const __bf16* __restrict__ T,        // [N][NSLOT][COUTP]
            const long long* __restrict__ ei,    // [2][E]
            const float* __restrict__ pseudo,    // [E]
            float* __restrict__ agg,             // [N][COUTA]
            int E) {
  using V = typename bfvec<CPL>::type;
  int gw = (int)((blockIdx.x * blockDim.x + threadIdx.x) >> 5);
  int lane = threadIdx.x & 31;
  if (gw >= E) return;
  int src = (int)ei[gw];
  int dst = (int)ei[(size_t)E + gw];
  float v = pseudo[gw] * (float)(KS - 1);
  float fb = floorf(v);
  float frac = v - fb;
  int i0 = (int)fb;
  i0 = i0 < 0 ? 0 : (i0 > KS - 1 ? KS - 1 : i0);
  int i1 = (i0 + 1 > KS - 1) ? KS - 1 : i0 + 1;
  int ch = lane * CPL;
  if (ch >= COUTA) return;
  V x0 = *(const V*)(T + ((size_t)src * NSLOT + i0) * COUTP + ch);
  V x1 = *(const V*)(T + ((size_t)src * NSLOT + i1) * COUTP + ch);
  float* ag = agg + (size_t)dst * COUTA + ch;
  float w0 = 1.0f - frac, w1 = frac;
#pragma unroll
  for (int j = 0; j < CPL; ++j) {
    float m = w0 * (float)x0[j] + w1 * (float)x1[j];
    atomicAdd(&ag[j], m);
  }
}

// ---------------- layer-1 combine: mean + root + bias + relu -> bf16 -------
__global__ void k_combine1(const float* __restrict__ agg,   // [N][128]
                           const float* __restrict__ deg,
                           const __bf16* __restrict__ T1,   // [N][NSLOT][128]
                           const float* __restrict__ b,
                           __bf16* __restrict__ hb, int N) {
  int i = blockIdx.x * blockDim.x + threadIdx.x;
  if (i >= N * 128) return;
  int n = i >> 7, c = i & 127;
  float d = deg[n];
  d = d < 1.0f ? 1.0f : d;
  float val = agg[i] / d + (float)T1[((size_t)n * NSLOT + (NSLOT - 1)) * 128 + c] + b[c];
  hb[i] = (__bf16)(val > 0.0f ? val : 0.0f);
}

// ---------------- layer-2 combine + log_softmax (warp per node) ------------
__global__ void k_combine2_lsm(const float* __restrict__ agg,  // [N][40]
                               const float* __restrict__ deg,
                               const __bf16* __restrict__ T2,  // [N][NSLOT][48]
                               const float* __restrict__ b,
                               float* __restrict__ out, int N) {
  int wave = (int)((blockIdx.x * blockDim.x + threadIdx.x) >> 5);
  int lane = threadIdx.x & 31;
  if (wave >= N) return;
  float d = deg[wave];
  d = d < 1.0f ? 1.0f : d;
  const float* ag = agg + (size_t)wave * 40;
  const __bf16* tr = T2 + ((size_t)wave * NSLOT + (NSLOT - 1)) * 48;
  const float NEG_INF = -__builtin_inff();
  int c0 = lane, c1 = lane + 32;
  float v0 = NEG_INF, v1 = NEG_INF;
  if (c0 < 40) v0 = ag[c0] / d + (float)tr[c0] + b[c0];
  if (c1 < 40) v1 = ag[c1] / d + (float)tr[c1] + b[c1];
  float m = fmaxf(v0, v1);
#pragma unroll
  for (int off = 16; off > 0; off >>= 1) m = fmaxf(m, __shfl_xor(m, off, 32));
  float s = 0.0f;
  if (c0 < 40) s += __expf(v0 - m);
  if (c1 < 40) s += __expf(v1 - m);
#pragma unroll
  for (int off = 16; off > 0; off >>= 1) s += __shfl_xor(s, off, 32);
  float lse = m + __logf(s);
  if (c0 < 40) out[(size_t)wave * 40 + c0] = v0 - lse;
  if (c1 < 40) out[(size_t)wave * 40 + c1] = v1 - lse;
}

// ---------------------------------------------------------------------------
extern "C" void kernel_launch(void* const* d_in, const int* in_sizes, int n_in,
                              void* d_out, int out_size, void* d_ws, size_t ws_size,
                              hipStream_t stream) {
  const float* x        = (const float*)d_in[0];
  const long long* ei   = (const long long*)d_in[1];   // int64 edge_index [2][E]
  const float* pseudo   = (const float*)d_in[2];       // [E][1]
  const float* W1       = (const float*)d_in[3];       // [8][64][128]
  const float* root1    = (const float*)d_in[4];       // [64][128]
  const float* b1       = (const float*)d_in[5];       // [128]
  const float* W2       = (const float*)d_in[6];       // [8][128][40]
  const float* root2    = (const float*)d_in[7];       // [128][40]
  const float* b2       = (const float*)d_in[8];       // [40]

  const int N = in_sizes[0] / 64;
  const int E = in_sizes[1] / 2;

  // --- carve workspace ---
  size_t off = 0;
  auto carve = [&](size_t bytes) -> void* {
    void* p = (char*)d_ws + off;
    off += (bytes + 255) & ~(size_t)255;
    return p;
  };
  __bf16* xb  = (__bf16*)carve((size_t)N * 64 * 2);
  __bf16* hb  = (__bf16*)carve((size_t)N * 128 * 2);
  __bf16* t1  = (__bf16*)carve((size_t)N * NSLOT * 128 * 2);
  __bf16* t2  = (__bf16*)carve((size_t)N * NSLOT * 48 * 2);
  float*  agg1 = (float*)carve((size_t)N * 128 * 4);
  float*  agg2 = (float*)carve((size_t)N * 40 * 4);
  float*  deg  = (float*)carve((size_t)N * 4);
  __bf16* w1p = (__bf16*)carve((size_t)NSLOT * 8 * 2 * 32 * 16 * 2); // [9][8ct][2ch][32][16]
  __bf16* w2p = (__bf16*)carve((size_t)NSLOT * 3 * 4 * 32 * 16 * 2); // [9][3ct][4ch][32][16]

  const int BLK = 256;
  auto cdiv = [](long long a, long long b) { return (int)((a + b - 1) / b); };

  // --- zero accumulators (capture-legal async memset) ---
  hipMemsetAsync(agg1, 0, (size_t)N * 128 * 4, stream);
  hipMemsetAsync(agg2, 0, (size_t)N * 40 * 4, stream);
  hipMemsetAsync(deg,  0, (size_t)N * 4, stream);

  // --- prep: bf16 convert + weight packing + degrees ---
  k_cvt_bf16<<<cdiv((long long)N * 64, BLK), BLK, 0, stream>>>(x, xb, N * 64);
  k_pack_w<64, 128, 128><<<cdiv(NSLOT * 8 * 2 * 32, BLK), BLK, 0, stream>>>(W1, root1, w1p);
  k_pack_w<128, 48, 40><<<cdiv(NSLOT * 3 * 4 * 32, BLK), BLK, 0, stream>>>(W2, root2, w2p);
  k_degree<<<cdiv(E, BLK), BLK, 0, stream>>>(ei, deg, E);

  const int rowTiles = (N + 15) / 16;

  // --- layer 1 ---
  k_transform<64, 128><<<cdiv((long long)rowTiles * NSLOT * 32, BLK), BLK, 0, stream>>>(
      xb, w1p, t1, N, rowTiles);
  k_edge<128, 128, 4><<<cdiv((long long)E * 32, BLK), BLK, 0, stream>>>(
      t1, ei, pseudo, agg1, E);
  k_combine1<<<cdiv((long long)N * 128, BLK), BLK, 0, stream>>>(
      agg1, deg, t1, b1, hb, N);

  // --- layer 2 ---
  k_transform<128, 48><<<cdiv((long long)rowTiles * NSLOT * 32, BLK), BLK, 0, stream>>>(
      hb, w2p, t2, N, rowTiles);
  k_edge<48, 40, 2><<<cdiv((long long)E * 32, BLK), BLK, 0, stream>>>(
      t2, ei, pseudo, agg2, E);
  k_combine2_lsm<<<cdiv((long long)N * 32, BLK), BLK, 0, stream>>>(
      agg2, deg, t2, b2, (float*)d_out, N);
}
